// Attention_15238543966444
// MI455X (gfx1250) — matrix operand order repaired and checked by reading, not compiled, not run on hardware
//
#include <hip/hip_runtime.h>

// ---------- types ----------
typedef __attribute__((ext_vector_type(16))) __bf16 v16bf;
typedef __attribute__((ext_vector_type(8)))  float  v8f;
typedef __attribute__((__vector_size__(16))) int    vint4;

#define AS1 __attribute__((address_space(1)))
#define AS3 __attribute__((address_space(3)))

// CDNA5 async global->LDS copy (ASYNCcnt-tracked), if the toolchain exposes it.
#if defined(__has_builtin)
#  if __has_builtin(__builtin_amdgcn_global_load_async_to_lds_b128)
#    define HAVE_ASYNC_LDS 1
#  endif
#endif

#ifdef HAVE_ASYNC_LDS
#  define ASYNC_COPY_B128(gp, lp)                                              \
      __builtin_amdgcn_global_load_async_to_lds_b128(                          \
          (AS1 vint4*)(AS1 void*)(void*)(gp),                                  \
          (AS3 vint4*)(AS3 void*)(void*)(lp), 0, 0)
#  if __has_builtin(__builtin_amdgcn_s_wait_asynccnt)
#    define WAIT_ASYNC() __builtin_amdgcn_s_wait_asynccnt(0)
#  else
#    define WAIT_ASYNC() asm volatile("s_wait_asynccnt 0x0" ::: "memory")
#  endif
#else
#  define WAIT_ASYNC()
#endif

union FragBF { v16bf v; uint4 q[2]; unsigned short u[16]; };

struct Acc { v8f a[2][2]; };

__device__ __forceinline__ unsigned short f32_to_bf16(float f) {
    unsigned int u = __float_as_uint(f);
    u += 0x7FFFu + ((u >> 16) & 1u);   // round-to-nearest-even
    return (unsigned short)(u >> 16);
}

// ---------- WMMA GEMM core: 64x64 block tile, 4 waves (128 thr), K step 32 ----
// A: row-major MxK bf16 (A points at tile row origin), lda in elements
// B: row-major KxN bf16 (B points at tile col origin), ldb in elements
__device__ __forceinline__ void gemm_core(const unsigned short* __restrict__ A, int lda,
                                          const unsigned short* __restrict__ B, int ldb,
                                          int K,
                                          unsigned short* As, unsigned short* Bs,
                                          Acc& acc)
{
    const int t    = threadIdx.x;
    const int lane = t & 31;
    const int wave = t >> 5;
    const int wm   = wave >> 1;
    const int wn   = wave & 1;
    const int half = lane >> 4;
    const int l16  = lane & 15;

    const v8f vzero = {0.f,0.f,0.f,0.f,0.f,0.f,0.f,0.f};
    for (int mt = 0; mt < 2; ++mt)
        for (int nt = 0; nt < 2; ++nt)
            acc.a[mt][nt] = vzero;

    const int arow = t >> 1;          // [0,64)
    const int acol = (t & 1) * 16;    // {0,16}
    const int bk   = t >> 2;          // [0,32)
    const int bn   = (t & 3) * 16;    // {0,16,32,48}

    for (int k0 = 0; k0 < K; k0 += 32) {
        // --- stage A tile 64x32 (row-major) ---
        const unsigned short* gA = A + (size_t)arow * lda + (size_t)(k0 + acol);
        unsigned short*       lA = &As[arow * 32 + acol];
#ifdef HAVE_ASYNC_LDS
        ASYNC_COPY_B128(gA,     lA);        // 16B, no VGPR round-trip
        ASYNC_COPY_B128(gA + 8, lA + 8);
#else
        {
            const uint4* ga = (const uint4*)gA;
            uint4 a0 = ga[0];
            uint4 a1 = ga[1];
            *(uint4*)lA       = a0;
            *(uint4*)(lA + 8) = a1;
        }
#endif

        // --- stage B tile 32x64, stored transposed as Bs[n][k] ---
        union { uint4 q[2]; unsigned short u[16]; } bb;
        const uint4* gb = (const uint4*)(B + (size_t)(k0 + bk) * ldb + (size_t)bn);
        bb.q[0] = gb[0];
        bb.q[1] = gb[1];
        #pragma unroll
        for (int i = 0; i < 16; ++i)
            Bs[(bn + i) * 32 + bk] = bb.u[i];

        // prefetch next K slab of B while WMMAs run
        if (k0 + 32 < K)
            __builtin_prefetch(B + (size_t)(k0 + 32 + bk) * ldb + (size_t)bn, 0, 1);

        WAIT_ASYNC();
        __syncthreads();

        // --- fragments (ISA 7.12.2 layouts) ---
        FragBF afr[2], bfr[2];
        #pragma unroll
        for (int mt = 0; mt < 2; ++mt) {
            int row = wm * 32 + mt * 16 + l16;
            // A lane holds k in [8*half, +8) and [16+8*half, +8)
            afr[mt].q[0] = *(const uint4*)&As[row * 32 + 8 * half];
            afr[mt].q[1] = *(const uint4*)&As[row * 32 + 16 + 8 * half];
        }
        #pragma unroll
        for (int nt = 0; nt < 2; ++nt) {
            int col = wn * 32 + nt * 16 + l16;
            // B lane holds k in [8*half, +8) and [16+8*half, +8)
            bfr[nt].q[0] = *(const uint4*)&Bs[col * 32 + 8 * half];
            bfr[nt].q[1] = *(const uint4*)&Bs[col * 32 + 16 + 8 * half];
        }

        #pragma unroll
        for (int mt = 0; mt < 2; ++mt)
            #pragma unroll
            for (int nt = 0; nt < 2; ++nt)
                acc.a[mt][nt] = __builtin_amdgcn_wmma_f32_16x16x32_bf16(
                    false, afr[mt].v, false, bfr[nt].v,
                    (short)0, acc.a[mt][nt], false, false);

        __syncthreads();
    }
}

// ---------- generic batched GEMM kernel ----------
// grid: (N/64, M/64, batch); strides in elements; f32 output
template <bool HAS_BIAS>
__global__ void __launch_bounds__(128)
gemm_bf16_kernel(const unsigned short* __restrict__ A, long long sA, int lda,
                 const unsigned short* __restrict__ B, long long sB, int ldb,
                 float* __restrict__ C, long long sC, int ldc,
                 const float* __restrict__ bias, int K)
{
    __shared__ unsigned short As[64 * 32];
    __shared__ unsigned short Bs[64 * 32];

    const int m0 = blockIdx.y * 64;
    const int n0 = blockIdx.x * 64;
    const int z  = blockIdx.z;

    const unsigned short* Ap = A + (size_t)z * (size_t)sA + (size_t)m0 * lda;
    const unsigned short* Bp = B + (size_t)z * (size_t)sB + (size_t)n0;

    Acc acc;
    gemm_core(Ap, lda, Bp, ldb, K, As, Bs, acc);

    const int t = threadIdx.x, lane = t & 31, wave = t >> 5;
    const int wm = wave >> 1, wn = wave & 1, half = lane >> 4, l16 = lane & 15;

    #pragma unroll
    for (int mt = 0; mt < 2; ++mt)
        #pragma unroll
        for (int nt = 0; nt < 2; ++nt) {
            int col = n0 + wn * 32 + nt * 16 + l16;
            float bv = HAS_BIAS ? bias[col] : 0.f;
            float* cp = C + (size_t)z * (size_t)sC + (size_t)(m0 + wm * 32 + mt * 16 + 8 * half) * ldc + col;
            #pragma unroll
            for (int v = 0; v < 8; ++v)
                cp[(size_t)v * ldc] = acc.a[mt][nt][v] + bv;
        }
}

// ---------- attn (shared over r) x V specialized GEMM ----------
// z = bn*8 + h ; A = attnb[b][h] (512x512), B = vb[bn][h] (512x64)
// writes Ob[bn][i][h*64+d] bf16
__global__ void __launch_bounds__(128)
attn_v_kernel(const unsigned short* __restrict__ attnb,
              const unsigned short* __restrict__ vb,
              unsigned short* __restrict__ Ob)
{
    __shared__ unsigned short As[64 * 32];
    __shared__ unsigned short Bs[64 * 32];

    const int z  = blockIdx.z;
    const int bn = z >> 3;
    const int h  = z & 7;
    const int b  = bn >> 5;
    const int m0 = blockIdx.y * 64;

    const unsigned short* Ap = attnb + ((size_t)(b * 8 + h)) * 512 * 512 + (size_t)m0 * 512;
    const unsigned short* Bp = vb + (size_t)z * 512 * 64;

    Acc acc;
    gemm_core(Ap, 512, Bp, 64, 512, As, Bs, acc);

    const int t = threadIdx.x, lane = t & 31, wave = t >> 5;
    const int wm = wave >> 1, wn = wave & 1, half = lane >> 4, l16 = lane & 15;

    #pragma unroll
    for (int mt = 0; mt < 2; ++mt)
        #pragma unroll
        for (int nt = 0; nt < 2; ++nt) {
            int col = wn * 32 + nt * 16 + l16;   // d in [0,64)
            unsigned short* op = Ob + (size_t)bn * 512 * 512
                               + (size_t)(m0 + wm * 32 + mt * 16 + 8 * half) * 512
                               + h * 64 + col;
            #pragma unroll
            for (int v = 0; v < 8; ++v)
                op[(size_t)v * 512] = f32_to_bf16(acc.a[mt][nt][v]);
        }
}

// ---------- elementwise / prep kernels ----------
__global__ void convert_bf16_kernel(const float* __restrict__ src,
                                    unsigned short* __restrict__ dst, size_t n)
{
    size_t i = (size_t)blockIdx.x * 256 + threadIdx.x;
    if (i < n) dst[i] = f32_to_bf16(src[i]);
}

// src[R][Ccols] f32 -> dst[Ccols][R] bf16
__global__ void transpose_convert_kernel(const float* __restrict__ src,
                                         unsigned short* __restrict__ dst,
                                         int R, int Ccols)
{
    size_t i = (size_t)blockIdx.x * 256 + threadIdx.x;
    if (i < (size_t)R * Ccols) {
        int r = (int)(i / Ccols);
        int c = (int)(i % Ccols);
        dst[(size_t)c * R + r] = f32_to_bf16(src[i]);
    }
}

// depthwise conv over n (kernel 15, pad 7): out bf16 [bn][n][c]
__global__ void dwconv_kernel(const float* __restrict__ x,
                              const float* __restrict__ dw_w,
                              const float* __restrict__ dw_b,
                              unsigned short* __restrict__ hdwT)
{
    size_t idx = (size_t)blockIdx.x * 256 + threadIdx.x;   // 64*512*256
    int c  = (int)(idx & 255);
    int n  = (int)((idx >> 8) & 511);
    int bn = (int)(idx >> 17);
    float acc = dw_b[c];
    #pragma unroll
    for (int tp = 0; tp < 15; ++tp) {
        int nn = n + tp - 7;
        if (nn >= 0 && nn < 512)
            acc += x[((size_t)bn * 512 + nn) * 256 + c] * dw_w[c * 15 + tp];
    }
    hdwT[idx] = f32_to_bf16(acc);
}

// mask stats per b: m_any[b][j], has_rows[b][r] (0/1 float), inv_s[b]=rsqrt(#rows)
__global__ void __launch_bounds__(512)
mask_stats_kernel(const unsigned char* __restrict__ mask,
                  float* __restrict__ has_rows, float* __restrict__ inv_s,
                  unsigned char* __restrict__ m_any)
{
    int b = blockIdx.x;
    int t = threadIdx.x;
    const unsigned char* mb = mask + (size_t)b * 32 * 512;
    unsigned char anyv = 0;
    for (int r = 0; r < 32; ++r) anyv |= (mb[r * 512 + t] != 0);
    m_any[b * 512 + t] = anyv;
    __shared__ float cnt[32];
    if (t < 32) {
        unsigned char hr = 0;
        for (int n = 0; n < 512; ++n) hr |= (mb[t * 512 + n] != 0);
        has_rows[b * 32 + t] = (float)hr;
        cnt[t] = (float)hr;
    }
    __syncthreads();
    if (t == 0) {
        float s = 0.f;
        for (int r = 0; r < 32; ++r) s += cnt[r];
        inv_s[b] = rsqrtf(s);
    }
}

// q: add bias, rotary, gate by has_rows, write qb[b][h][i][r*64+d] bf16
__global__ void rotary_q_kernel(const float* __restrict__ qf,
                                const float* __restrict__ pw_b,
                                const float* __restrict__ has_rows,
                                unsigned short* __restrict__ qb)
{
    size_t idx = (size_t)blockIdx.x * 256 + threadIdx.x;   // 64*512*512
    int o  = (int)(idx & 511);
    int i  = (int)((idx >> 9) & 511);
    int bn = (int)(idx >> 18);
    int h = o >> 6, d = o & 63, r = bn & 31, b = bn >> 5;
    float v  = qf[idx]     + pw_b[o];
    float vp = qf[idx ^ 1] + pw_b[o ^ 1];
    int   t2   = d & ~1;
    float freq = __powf(10000.f, -(float)t2 * (1.f / 64.f));
    float ang  = (float)i * freq;
    float s = __sinf(ang), c = __cosf(ang);
    float out = (d & 1) ? (v * c + vp * s) : (v * c - vp * s);
    out *= has_rows[b * 32 + r];
    qb[((size_t)(b * 8 + h) * 512 + i) * 2048 + r * 64 + d] = f32_to_bf16(out);
}

// k: rotary, write kb[b][h][r*64+d][j] bf16 ; v: write vb[bn][h][j][d] bf16
__global__ void rotary_kv_kernel(const float* __restrict__ kvf,
                                 unsigned short* __restrict__ kb,
                                 unsigned short* __restrict__ vb)
{
    size_t idx = (size_t)blockIdx.x * 256 + threadIdx.x;   // 64*512*512
    int o  = (int)(idx & 511);
    int j  = (int)((idx >> 9) & 511);
    int bn = (int)(idx >> 18);
    int h = o >> 6, d = o & 63, r = bn & 31, b = bn >> 5;
    size_t base = ((size_t)bn * 512 + j) * 1024;
    float v  = kvf[base + o];
    float vp = kvf[base + (o ^ 1)];
    int   t2   = d & ~1;
    float freq = __powf(10000.f, -(float)t2 * (1.f / 64.f));
    float ang  = (float)j * freq;
    float s = __sinf(ang), c = __cosf(ang);
    float kout = (d & 1) ? (v * c + vp * s) : (v * c - vp * s);
    kb[((size_t)(b * 8 + h) * 2048 + r * 64 + d) * 512 + j] = f32_to_bf16(kout);
    float vv = kvf[base + 512 + o];
    vb[((size_t)bn * 8 + h) * 512 * 64 + (size_t)j * 64 + d] = f32_to_bf16(vv);
}

// softmax per row (b,h,i) over j=0..511, writes attnb bf16
__global__ void __launch_bounds__(256)
softmax_kernel(const float* __restrict__ dots,
               const unsigned char* __restrict__ m_any,
               const float* __restrict__ inv_s,
               unsigned short* __restrict__ attnb)
{
    const int row = blockIdx.x;          // (b*8+h)*512 + i
    const int i   = row & 511;
    const int b   = (row >> 9) >> 3;
    const float* dp = dots + (size_t)row * 512;
    const float scale = 0.125f * inv_s[b];
    const bool mi = (m_any[b * 512 + i] != 0);
    const float NEGMAX = -3.402823466e38f;

    __shared__ float red[256];
    int t = threadIdx.x;
    int j0 = t, j1 = t + 256;
    float l0 = (mi && m_any[b * 512 + j0]) ? dp[j0] * scale : NEGMAX;
    float l1 = (mi && m_any[b * 512 + j1]) ? dp[j1] * scale : NEGMAX;

    red[t] = fmaxf(l0, l1);
    __syncthreads();
    for (int s = 128; s > 0; s >>= 1) {
        if (t < s) red[t] = fmaxf(red[t], red[t + s]);
        __syncthreads();
    }
    float m = red[0];
    __syncthreads();

    float e0 = __expf(l0 - m);
    float e1 = __expf(l1 - m);
    red[t] = e0 + e1;
    __syncthreads();
    for (int s = 128; s > 0; s >>= 1) {
        if (t < s) red[t] += red[t + s];
        __syncthreads();
    }
    float inv = 1.f / red[0];

    attnb[(size_t)row * 512 + j0] = f32_to_bf16(e0 * inv);
    attnb[(size_t)row * 512 + j1] = f32_to_bf16(e1 * inv);
}

// ---------- host launch ----------
extern "C" void kernel_launch(void* const* d_in, const int* in_sizes, int n_in,
                              void* d_out, int out_size, void* d_ws, size_t ws_size,
                              hipStream_t stream)
{
    (void)in_sizes; (void)n_in; (void)out_size; (void)ws_size;

    const float*         x    = (const float*)d_in[0];
    const unsigned char* mask = (const unsigned char*)d_in[1];
    const float*         dw_w = (const float*)d_in[3];
    const float*         dw_b = (const float*)d_in[4];
    const float*         pw_w = (const float*)d_in[5];
    const float*         pw_b = (const float*)d_in[6];
    const float*         w_kv = (const float*)d_in[7];
    const float*         w_o  = (const float*)d_in[8];
    const float*         b_o  = (const float*)d_in[9];
    float* out = (float*)d_out;

    char* ws = (char*)d_ws;
    size_t off = 0;
    auto alloc = [&](size_t bytes) {
        size_t o = off;
        off += (bytes + 255) & ~(size_t)255;
        return o;
    };

    const size_t BN = 64, N = 512, C = 256, INNER = 512;

    unsigned short* xb     = (unsigned short*)(ws + alloc(BN * N * C * 2));          // x bf16
    unsigned short* hdwT   = (unsigned short*)(ws + alloc(BN * N * C * 2));          // conv out bf16
    unsigned short* pw_wT  = (unsigned short*)(ws + alloc(C * INNER * 2));           // (c,o)
    unsigned short* w_kvb  = (unsigned short*)(ws + alloc(C * 2 * INNER * 2));
    unsigned short* w_ob   = (unsigned short*)(ws + alloc(INNER * C * 2));
    float*          qf32   = (float*)(ws + alloc(BN * N * INNER * 4));
    float*          kvf32  = (float*)(ws + alloc(BN * N * 2 * INNER * 4));
    unsigned short* qb     = (unsigned short*)(ws + alloc((size_t)16 * 512 * 2048 * 2)); // [b*8+h][i][rd]
    unsigned short* kb     = (unsigned short*)(ws + alloc((size_t)16 * 2048 * 512 * 2)); // [b*8+h][rd][j]
    unsigned short* vb     = (unsigned short*)(ws + alloc(BN * 8 * 512 * 64 * 2));       // [bn][h][j][d]
    float*          dots   = (float*)(ws + alloc((size_t)16 * 512 * 512 * 4));
    unsigned short* attnb  = (unsigned short*)(ws + alloc((size_t)16 * 512 * 512 * 2));
    unsigned short* Ob     = (unsigned short*)(ws + alloc(BN * N * INNER * 2));          // [bn][i][h*64+d]
    float*          has_rows = (float*)(ws + alloc(64 * 4));
    float*          inv_s    = (float*)(ws + alloc(2 * 4));
    unsigned char*  m_any    = (unsigned char*)(ws + alloc(2 * 512));

    // 1) conversions / weight prep
    {
        size_t n = BN * N * C;
        convert_bf16_kernel<<<dim3((n + 255) / 256), dim3(256), 0, stream>>>(x, xb, n);
    }
    transpose_convert_kernel<<<dim3((INNER * C + 255) / 256), dim3(256), 0, stream>>>(pw_w, pw_wT, (int)INNER, (int)C);
    {
        size_t n = C * 2 * INNER;
        convert_bf16_kernel<<<dim3((n + 255) / 256), dim3(256), 0, stream>>>(w_kv, w_kvb, n);
    }
    {
        size_t n = INNER * C;
        convert_bf16_kernel<<<dim3((n + 255) / 256), dim3(256), 0, stream>>>(w_o, w_ob, n);
    }

    // 2) mask stats
    mask_stats_kernel<<<dim3(2), dim3(512), 0, stream>>>(mask, has_rows, inv_s, m_any);

    // 3) depthwise conv
    dwconv_kernel<<<dim3((BN * N * C) / 256), dim3(256), 0, stream>>>(x, dw_w, dw_b, hdwT);

    // 4) q = hdwT (512x256) x pw_wT (256x512) per bn  -> qf32
    gemm_bf16_kernel<false><<<dim3(8, 8, 64), dim3(128), 0, stream>>>(
        hdwT, (long long)(N * C), (int)C,
        pw_wT, 0LL, (int)INNER,
        qf32, (long long)(N * INNER), (int)INNER,
        nullptr, (int)C);

    // 5) kv = xb (512x256) x w_kvb (256x1024) per bn -> kvf32
    gemm_bf16_kernel<false><<<dim3(16, 8, 64), dim3(128), 0, stream>>>(
        xb, (long long)(N * C), (int)C,
        w_kvb, 0LL, (int)(2 * INNER),
        kvf32, (long long)(N * 2 * INNER), (int)(2 * INNER),
        nullptr, (int)C);

    // 6) rotary + layout shuffles
    rotary_q_kernel<<<dim3((BN * N * INNER) / 256), dim3(256), 0, stream>>>(qf32, pw_b, has_rows, qb);
    rotary_kv_kernel<<<dim3((BN * N * INNER) / 256), dim3(256), 0, stream>>>(kvf32, kb, vb);

    // 7) dots = qb (512x2048) x kb (2048x512) per (b,h) -> dots f32
    gemm_bf16_kernel<false><<<dim3(8, 8, 16), dim3(128), 0, stream>>>(
        qb, (long long)(512 * 2048), 2048,
        kb, (long long)(2048 * 512), 512,
        dots, (long long)(512 * 512), 512,
        nullptr, 2048);

    // 8) masked softmax -> attnb bf16
    softmax_kernel<<<dim3(16 * 512), dim3(256), 0, stream>>>(dots, m_any, inv_s, attnb);

    // 9) out = attn (512x512) x v (512x64) per (bn,h) -> Ob bf16 [bn][i][h*64+d]
    attn_v_kernel<<<dim3(1, 8, 512), dim3(128), 0, stream>>>(attnb, vb, Ob);

    // 10) y = Ob (512x512) x w_ob (512x256) + b_o per bn -> d_out f32
    gemm_bf16_kernel<true><<<dim3(4, 8, 64), dim3(128), 0, stream>>>(
        Ob, (long long)(N * INNER), (int)INNER,
        w_ob, 0LL, (int)C,
        out, (long long)(N * C), (int)C,
        b_o, (int)INNER);
}
